// Concate_GCN_4518305595467
// MI455X (gfx1250) — matrix operand order repaired and checked by reading, not compiled, not run on hardware
//
#include <hip/hip_runtime.h>
#include <hip/hip_bf16.h>
#include <stdint.h>

// ---------------------------------------------------------------------------
// CDNA5 / gfx1250 Concate_GCN forward pass.
// GEMMs: v_wmma_f32_16x16x32_bf16 (wave32). Tile copies: async global->LDS.
// LDS tiles padded to ld=72 elements (conflict-free b128 fragment loads).
// ---------------------------------------------------------------------------

typedef __bf16 bf16_t;
typedef __attribute__((ext_vector_type(16))) __bf16 v16bf;
typedef __attribute__((ext_vector_type(8)))  __bf16 v8bf;
typedef __attribute__((ext_vector_type(8)))  float  v8f;
typedef __attribute__((ext_vector_type(4)))  int    v4i_t;

#define NGC 640
#define NAF 25
#define LDT 72   // padded leading dim for 64-wide LDS tiles

#if defined(__has_builtin)
#if __has_builtin(__builtin_amdgcn_global_load_async_to_lds_b128) && \
    __has_builtin(__builtin_amdgcn_s_wait_asynccnt)
#define HAVE_ASYNC_LDS 1
#endif
#endif

typedef __attribute__((address_space(1))) v4i_t* gptr_v4i;
typedef __attribute__((address_space(3))) v4i_t* lptr_v4i;

struct b128_t { uint32_t v[4]; };

__device__ __forceinline__ void async_copy_b128(const void* g, void* lds) {
#ifdef HAVE_ASYNC_LDS
  __builtin_amdgcn_global_load_async_to_lds_b128(
      (gptr_v4i)const_cast<void*>(g), (lptr_v4i)lds, 0, 0);
#else
  *(b128_t*)lds = *(const b128_t*)g;
#endif
}

__device__ __forceinline__ void async_wait() {
#ifdef HAVE_ASYNC_LDS
  __builtin_amdgcn_s_wait_asynccnt(0);
#endif
}

// ---------------- WMMA helpers ----------------

__device__ __forceinline__ v8f wmma_bf16(v16bf a, v16bf b, v8f c) {
  return __builtin_amdgcn_wmma_f32_16x16x32_bf16(false, a, false, b, (short)0, c,
                                                 false, false);
}

__device__ __forceinline__ v8f zero_v8f() {
  v8f z;
#pragma unroll
  for (int i = 0; i < 8; ++i) z[i] = 0.0f;
  return z;
}

// A-fragment 16x32 (M x K) from row-major LDS tile (ld multiple of 8).
// Per lane: two contiguous 16B runs -> two ds_load_b128.
__device__ __forceinline__ v16bf load_a_frag(const bf16_t* src, int ld, int row0, int k0) {
  int lane = threadIdx.x & 31;
  int l = lane & 15, hi = lane >> 4;
  const bf16_t* r = src + (row0 + l) * ld + k0 + hi * 16;
  v8bf lo = *(const v8bf*)r;
  v8bf up = *(const v8bf*)(r + 8);
  return __builtin_shufflevector(lo, up, 0, 1, 2, 3, 4, 5, 6, 7,
                                 8, 9, 10, 11, 12, 13, 14, 15);
}

// B-fragment 32x16 (K x N) from *transposed* (N-major) LDS tile.
// Per lane: 16 contiguous bf16 -> two ds_load_b128.
__device__ __forceinline__ v16bf load_bt_frag(const bf16_t* srcT, int ldt, int k0, int n0) {
  int lane = threadIdx.x & 31;
  int n = lane & 15, hi = lane >> 4;
  const bf16_t* r = srcT + (n0 + n) * ldt + k0 + hi * 16;
  v8bf lo = *(const v8bf*)r;
  v8bf up = *(const v8bf*)(r + 8);
  return __builtin_shufflevector(lo, up, 0, 1, 2, 3, 4, 5, 6, 7,
                                 8, 9, 10, 11, 12, 13, 14, 15);
}

// ---------------- small prep kernels ----------------

__global__ void mb_kernel(const float* __restrict__ adjs, float* __restrict__ mb) {
  int idx = blockIdx.x * blockDim.x + threadIdx.x;  // b*64+n over 32768
  const float* row = adjs + (size_t)idx * 64;
  float m = 0.0f;
#pragma unroll 8
  for (int k = 0; k < 64; ++k) m = fmaxf(m, row[k]);
  mb[idx] = m;
}

// atom features, transposed + padded: out [B][32 feat][64 atoms] bf16
__global__ void afm_prep_kernel(const float* __restrict__ afms, bf16_t* __restrict__ out) {
  int idx = blockIdx.x * blockDim.x + threadIdx.x;  // B*32*64
  int n = idx & 63;
  int c = (idx >> 6) & 31;
  int b = idx >> 11;
  float v = (c < NAF) ? afms[((size_t)b * 64 + n) * NAF + c] : 0.0f;
  out[idx] = (bf16_t)v;
}

__global__ void concat5_kernel(const float* __restrict__ p0, const float* __restrict__ p1,
                               const float* __restrict__ p2, const float* __restrict__ p3,
                               const float* __restrict__ p4, float* __restrict__ out) {
  int i = blockIdx.x * blockDim.x + threadIdx.x;
  if (i >= NGC) return;
  const float* p = (i < 128) ? p0 : (i < 256) ? p1 : (i < 384) ? p2 : (i < 512) ? p3 : p4;
  out[i] = p[i & 127];
}

// ---------------- attention ----------------

__device__ __forceinline__ float bred_max64(float v, float* buf) {
  int t = threadIdx.x;
  buf[t] = v; __syncthreads();
  for (int s = 32; s > 0; s >>= 1) {
    if (t < s) buf[t] = fmaxf(buf[t], buf[t + s]);
    __syncthreads();
  }
  float r = buf[0]; __syncthreads();
  return r;
}

__device__ __forceinline__ float bred_sum64(float v, float* buf) {
  int t = threadIdx.x;
  buf[t] = v; __syncthreads();
  for (int s = 32; s > 0; s >>= 1) {
    if (t < s) buf[t] += buf[t + s];
    __syncthreads();
  }
  float r = buf[0]; __syncthreads();
  return r;
}

// grid (64 rows, 512 batches), block 64
__global__ void attention_kernel(const float* __restrict__ rel, int C,
                                 const float* __restrict__ w,
                                 const float* __restrict__ adjs,
                                 bf16_t* __restrict__ Aout, int dbl) {
  __shared__ float buf[64];
  int t = threadIdx.x;
  int row = blockIdx.x, b = blockIdx.y;
  const float* adjRow = adjs + ((size_t)b * 64 + row) * 64;
  float aval = adjRow[t];
  float mask = (1.0f - aval) * (-1.0e9f);
  float s = 0.0f;
  for (int c = 0; c < C; ++c)
    s += w[c] * rel[(((size_t)b * C + c) * 64 + row) * 64 + t];
  float mb = bred_max64(aval, buf);
  float x  = s + mask;
  float mx = bred_max64(x, buf);
  float e  = __expf(x - mx);
  float sm = bred_sum64(e, buf);
  float A  = e / sm * mb;
  for (int it = 0; it < dbl; ++it) {
    x  = A + mask;
    mx = bred_max64(x, buf);
    e  = __expf(x - mx);
    sm = bred_sum64(e, buf);
    A  = e / sm * mb;
  }
  Aout[((size_t)b * 64 + row) * 64 + t] = (bf16_t)A;
}

// ---------------- layer-1 graph conv: H = (A @ afm) @ W1 + b ----------------
__global__ void gconv1_kernel(const bf16_t* __restrict__ Arel,  // [B,64,64] bf16
                              const bf16_t* __restrict__ XaT,   // [B,32,64] bf16 (transposed)
                              const float*  __restrict__ W,     // [25,128] f32
                              const float*  __restrict__ bias,  // [128]
                              float* __restrict__ H, int colOff) {
  __shared__ __attribute__((aligned(16))) bf16_t sA[64 * LDT];
  __shared__ __attribute__((aligned(16))) bf16_t sXt[32 * LDT];  // feat-major afm
  __shared__ __attribute__((aligned(16))) bf16_t sWt[128 * 40];  // n-major W, ld=40
  __shared__ __attribute__((aligned(16))) bf16_t sT[64 * 40];    // T, row-major, ld=40
  int b = blockIdx.x;
  int t = threadIdx.x, wave = t >> 5, lane = t & 31, hi = lane >> 4;

  // async tile copies (b128 granularity, 16B both sides)
  for (int i = t; i < 512; i += 256) {          // A: 64 rows x 8 chunks
    int r = i >> 3, ch = i & 7;
    async_copy_b128(Arel + (size_t)b * 4096 + r * 64 + ch * 8, sA + r * LDT + ch * 8);
  }
  for (int i = t; i < 256; i += 256) {          // afm^T: 32 rows x 8 chunks
    int r = i >> 3, ch = i & 7;
    async_copy_b128(XaT + ((size_t)b * 32 + r) * 64 + ch * 8, sXt + r * LDT + ch * 8);
  }
  // W1 -> transposed bf16 (overlaps with async DMA above)
  for (int i = t; i < 32 * 128; i += 256) {
    int k = i >> 7, c = i & 127;
    sWt[c * 40 + k] = (bf16_t)((k < NAF) ? W[k * 128 + c] : 0.0f);
  }
  async_wait();
  __syncthreads();

  // T = A @ afm : 64x32, K=64 -> 8 tiles, one per wave
  {
    int tm = wave >> 1, tn = wave & 1;
    v8f c = zero_v8f();
    c = wmma_bf16(load_a_frag(sA, LDT, tm * 16, 0),  load_bt_frag(sXt, LDT, 0,  tn * 16), c);
    c = wmma_bf16(load_a_frag(sA, LDT, tm * 16, 32), load_bt_frag(sXt, LDT, 32, tn * 16), c);
#pragma unroll
    for (int r = 0; r < 8; ++r)
      sT[(tm * 16 + r + 8 * hi) * 40 + tn * 16 + (lane & 15)] = (bf16_t)c[r];
  }
  __syncthreads();

  // H = T @ W : 64x128, K=32 -> wave owns one 16-wide column tile
  v16bf bf = load_bt_frag(sWt, 40, 0, wave * 16);
#pragma unroll
  for (int m = 0; m < 4; ++m) {
    v8f c = zero_v8f();
    c = wmma_bf16(load_a_frag(sT, 40, m * 16, 0), bf, c);
#pragma unroll
    for (int r = 0; r < 8; ++r) {
      int row = m * 16 + r + 8 * hi;
      int col = wave * 16 + (lane & 15);
      H[((size_t)b * 64 + row) * NGC + colOff + col] = c[r] + bias[col];
    }
  }
}

// ---------------- layer-2 graph conv: H = (A @ x1) @ W2 + b (fused) ---------
__global__ void gconv2_kernel(const bf16_t* __restrict__ Arel,  // [B,64,64] bf16
                              const bf16_t* __restrict__ X1T,   // [B,640,64] bf16 (feat-major)
                              const float*  __restrict__ W,     // [640,128]
                              const float*  __restrict__ bias,  // [128]
                              float* __restrict__ H, int colOff) {
  __shared__ __attribute__((aligned(16))) bf16_t sA[64 * LDT];
  __shared__ __attribute__((aligned(16))) bf16_t sXt[64 * LDT];  // 64 feats x 64 atoms
  __shared__ __attribute__((aligned(16))) bf16_t sT[64 * LDT];   // atoms x feats
  __shared__ __attribute__((aligned(16))) bf16_t sWt[128 * LDT]; // n-major W chunk
  int b = blockIdx.x;
  int t = threadIdx.x, wave = t >> 5, lane = t & 31, hi = lane >> 4;

  for (int i = t; i < 512; i += 256) {   // attention tile, async DMA
    int r = i >> 3, ch = i & 7;
    async_copy_b128(Arel + (size_t)b * 4096 + r * 64 + ch * 8, sA + r * LDT + ch * 8);
  }

  v8f acc[4];
#pragma unroll
  for (int m = 0; m < 4; ++m) acc[m] = zero_v8f();

  for (int k0 = 0; k0 < NGC; k0 += 64) {
    // x1^T chunk: 64 feature rows x 64 atoms, async DMA
    for (int i = t; i < 512; i += 256) {
      int r = i >> 3, ch = i & 7;
      async_copy_b128(X1T + ((size_t)b * NGC + k0 + r) * 64 + ch * 8,
                      sXt + r * LDT + ch * 8);
    }
    if (k0 + 64 < NGC)
      __builtin_prefetch(&W[(size_t)(k0 + 64) * 128 + (t & 127)], 0, 1);
    // W chunk -> transposed bf16 (overlaps DMA)
    for (int i = t; i < 8192; i += 256) {
      int k = i >> 7, c = i & 127;
      sWt[c * LDT + k] = (bf16_t)W[(size_t)(k0 + k) * 128 + c];
    }
    async_wait();
    __syncthreads();

    // T = A @ Xchunk : 64x64, K=64 -> 16 tiles, 2 per wave
    {
      int tm = wave >> 1, tn0 = (wave & 1) * 2;
#pragma unroll
      for (int j = 0; j < 2; ++j) {
        int tn = tn0 + j;
        v8f c = zero_v8f();
        c = wmma_bf16(load_a_frag(sA, LDT, tm * 16, 0),  load_bt_frag(sXt, LDT, 0,  tn * 16), c);
        c = wmma_bf16(load_a_frag(sA, LDT, tm * 16, 32), load_bt_frag(sXt, LDT, 32, tn * 16), c);
#pragma unroll
        for (int r = 0; r < 8; ++r)
          sT[(tm * 16 + r + 8 * hi) * LDT + tn * 16 + (lane & 15)] = (bf16_t)c[r];
      }
    }
    __syncthreads();

    // acc += T @ Wchunk : 64x128, K=64
    v16bf b0 = load_bt_frag(sWt, LDT, 0,  wave * 16);
    v16bf b1 = load_bt_frag(sWt, LDT, 32, wave * 16);
#pragma unroll
    for (int m = 0; m < 4; ++m) {
      acc[m] = wmma_bf16(load_a_frag(sT, LDT, m * 16, 0),  b0, acc[m]);
      acc[m] = wmma_bf16(load_a_frag(sT, LDT, m * 16, 32), b1, acc[m]);
    }
    __syncthreads();
  }
#pragma unroll
  for (int m = 0; m < 4; ++m)
#pragma unroll
    for (int r = 0; r < 8; ++r) {
      int row = m * 16 + r + 8 * hi;
      int col = wave * 16 + (lane & 15);
      H[((size_t)b * 64 + row) * NGC + colOff + col] = acc[m][r] + bias[col];
    }
}

// ---------------- BatchNorm over (B,N) rows ----------------

__global__ void bn_stats_kernel(const float* __restrict__ H, int rows, int C,
                                float* __restrict__ mean, float* __restrict__ rstd) {
  __shared__ float s1[256], s2[256];
  int c = blockIdx.x, t = threadIdx.x;
  float a = 0.0f, b = 0.0f;
  for (int r = t; r < rows; r += 256) {
    float v = H[(size_t)r * C + c];
    a += v; b += v * v;
  }
  s1[t] = a; s2[t] = b; __syncthreads();
  for (int s = 128; s > 0; s >>= 1) {
    if (t < s) { s1[t] += s1[t + s]; s2[t] += s2[t + s]; }
    __syncthreads();
  }
  if (t == 0) {
    float m = s1[0] / rows;
    float v = s2[0] / rows - m * m;
    mean[c] = m;
    rstd[c] = rsqrtf(v + 1e-5f);
  }
}

// BN + relu + row-mask -> bf16, written TRANSPOSED per batch: x1t [B][C][64]
__global__ void bn_apply_mask_kernel(const float* __restrict__ H,
                                     const float* __restrict__ mean,
                                     const float* __restrict__ rstd,
                                     const float* __restrict__ g,
                                     const float* __restrict__ bb,
                                     const float* __restrict__ mb,
                                     bf16_t* __restrict__ out, int C) {
  int idx = blockIdx.x * 256 + threadIdx.x;   // output index: (b*C + c)*64 + n
  int n = idx & 63;
  int bc = idx >> 6;
  int c = bc % C;
  int b = bc / C;
  float v = g[c] * (H[((size_t)b * 64 + n) * C + c] - mean[c]) * rstd[c] + bb[c];
  v = fmaxf(v, 0.0f) * mb[b * 64 + n];
  out[idx] = (bf16_t)v;
}

// BN + relu + row-mask + sum over atoms -> MolFP [B,C]
__global__ void bn_apply_sum_kernel(const float* __restrict__ H,
                                    const float* __restrict__ mean,
                                    const float* __restrict__ rstd,
                                    const float* __restrict__ g,
                                    const float* __restrict__ bb,
                                    const float* __restrict__ mb,
                                    float* __restrict__ molfp, int C) {
  int idx = blockIdx.x * 256 + threadIdx.x;  // b*C + c
  int c = idx % C, b = idx / C;
  float m = mean[c], rs = rstd[c], gg = g[c], bbc = bb[c];
  float acc = 0.0f;
#pragma unroll 4
  for (int n = 0; n < 64; ++n) {
    float v = gg * (H[((size_t)b * 64 + n) * C + c] - m) * rs + bbc;
    acc += fmaxf(v, 0.0f) * mb[b * 64 + n];
  }
  molfp[idx] = acc;
}

// BN over batch dim (R rows), optional relu, emit bf16
__global__ void bn1d_kernel(const float* __restrict__ X,
                            const float* __restrict__ g, const float* __restrict__ bb,
                            bf16_t* __restrict__ out, int R, int C, int relu) {
  __shared__ float s1[256], s2[256];
  int c = blockIdx.x, t = threadIdx.x;
  float a = 0.0f, b2 = 0.0f;
  for (int r = t; r < R; r += 256) {
    float v = X[(size_t)r * C + c];
    a += v; b2 += v * v;
  }
  s1[t] = a; s2[t] = b2; __syncthreads();
  for (int s = 128; s > 0; s >>= 1) {
    if (t < s) { s1[t] += s1[t + s]; s2[t] += s2[t + s]; }
    __syncthreads();
  }
  float m  = s1[0] / R;
  float rs = rsqrtf(s2[0] / R - m * m + 1e-5f);
  __syncthreads();
  for (int r = t; r < R; r += 256) {
    float v = g[c] * (X[(size_t)r * C + c] - m) * rs + bb[c];
    if (relu) v = fmaxf(v, 0.0f);
    out[(size_t)r * C + c] = (bf16_t)v;
  }
}

// ---------------- dense GEMM: out = Abf16[M,K] @ Wf32[K,N] + bias -----------
// block 256 threads, tile 64(M) x 128(N), K multiple of 64
__global__ void gemm_kernel(const bf16_t* __restrict__ A, const float* __restrict__ W,
                            const float* __restrict__ bias, float* __restrict__ out,
                            int N, int K) {
  __shared__ __attribute__((aligned(16))) bf16_t sA[64 * LDT];
  __shared__ __attribute__((aligned(16))) bf16_t sWt[128 * LDT];
  int n0 = blockIdx.x * 128, m0 = blockIdx.y * 64;
  int t = threadIdx.x, wave = t >> 5, lane = t & 31, hi = lane >> 4;
  v8f acc[4];
#pragma unroll
  for (int m = 0; m < 4; ++m) acc[m] = zero_v8f();
  for (int k0 = 0; k0 < K; k0 += 64) {
    for (int i = t; i < 512; i += 256) {    // A tile, async DMA
      int r = i >> 3, ch = i & 7;
      async_copy_b128(A + (size_t)(m0 + r) * K + k0 + ch * 8, sA + r * LDT + ch * 8);
    }
    if (k0 + 64 < K)
      __builtin_prefetch(&W[(size_t)(k0 + 64) * N + n0 + (t & 127)], 0, 1);
    for (int i = t; i < 8192; i += 256) {   // W tile -> transposed bf16
      int k = i >> 7, c = i & 127;
      sWt[c * LDT + k] = (bf16_t)W[(size_t)(k0 + k) * N + n0 + c];
    }
    async_wait();
    __syncthreads();
    v16bf b0 = load_bt_frag(sWt, LDT, 0,  wave * 16);
    v16bf b1 = load_bt_frag(sWt, LDT, 32, wave * 16);
#pragma unroll
    for (int m = 0; m < 4; ++m) {
      acc[m] = wmma_bf16(load_a_frag(sA, LDT, m * 16, 0),  b0, acc[m]);
      acc[m] = wmma_bf16(load_a_frag(sA, LDT, m * 16, 32), b1, acc[m]);
    }
    __syncthreads();
  }
#pragma unroll
  for (int m = 0; m < 4; ++m)
#pragma unroll
    for (int r = 0; r < 8; ++r) {
      int row = m * 16 + r + 8 * hi;
      int col = wave * 16 + (lane & 15);
      out[(size_t)(m0 + row) * N + n0 + col] = acc[m][r] + bias[n0 + col];
    }
}

// ---------------- final dense: [512,256] @ [256,1] + b ----------------------
__global__ void dense3_kernel(const bf16_t* __restrict__ X, const float* __restrict__ W,
                              const float* __restrict__ b3, float* __restrict__ out) {
  __shared__ float s1[256];
  int b = blockIdx.x, t = threadIdx.x;
  s1[t] = (float)X[(size_t)b * 256 + t] * W[t];
  __syncthreads();
  for (int s = 128; s > 0; s >>= 1) {
    if (t < s) s1[t] += s1[t + s];
    __syncthreads();
  }
  if (t == 0) out[b] = s1[0] + b3[0];
}

// ---------------------------------------------------------------------------

extern "C" void kernel_launch(void* const* d_in, const int* in_sizes, int n_in,
                              void* d_out, int out_size, void* d_ws, size_t ws_size,
                              hipStream_t stream) {
  (void)in_sizes; (void)n_in; (void)out_size; (void)ws_size;

  const float* adjs = (const float*)d_in[0];
  const float* afms = (const float*)d_in[1];
  const float* rels[5] = {(const float*)d_in[2], (const float*)d_in[3],
                          (const float*)d_in[4], (const float*)d_in[5],
                          (const float*)d_in[6]};
  const int relC[5] = {25, 5, 3, 3, 3};

  char* ws = (char*)d_ws;
  size_t off = 0;
  auto alloc = [&](size_t bytes) -> void* {
    void* p = ws + off;
    off = (off + bytes + 255) & ~(size_t)255;
    return p;
  };
  bf16_t* Abf   = (bf16_t*)alloc(5ull * 512 * 4096 * 2);   // attentions (reused L1/L2)
  bf16_t* afmT  = (bf16_t*)alloc(512ull * 32 * 64 * 2);    // afm, feature-major
  bf16_t* x1t   = (bf16_t*)alloc(512ull * NGC * 64 * 2);   // x1, feature-major
  float*  H     = (float*) alloc(32768ull * NGC * 4);      // pre-BN H1, then H2
  float*  mbuf  = (float*) alloc(32768ull * 4);
  float*  mean  = (float*) alloc(NGC * 4);
  float*  rstd  = (float*) alloc(NGC * 4);
  float*  g1    = (float*) alloc(NGC * 4);
  float*  b1c   = (float*) alloc(NGC * 4);
  float*  g2    = (float*) alloc(NGC * 4);
  float*  b2c   = (float*) alloc(NGC * 4);
  float*  molfp = (float*) alloc(512ull * NGC * 4);
  bf16_t* xb0   = (bf16_t*)alloc(512ull * NGC * 2);
  float*  y1    = (float*) alloc(512ull * 512 * 4);
  bf16_t* xb1   = (bf16_t*)alloc(512ull * 512 * 2);
  float*  y2    = (float*) alloc(512ull * 256 * 4);
  bf16_t* xb2   = (bf16_t*)alloc(512ull * 256 * 2);

  // prep
  mb_kernel<<<128, 256, 0, stream>>>(adjs, mbuf);
  afm_prep_kernel<<<4096, 256, 0, stream>>>(afms, afmT);
  concat5_kernel<<<3, 256, 0, stream>>>((const float*)d_in[37], (const float*)d_in[38],
                                        (const float*)d_in[39], (const float*)d_in[40],
                                        (const float*)d_in[41], g1);
  concat5_kernel<<<3, 256, 0, stream>>>((const float*)d_in[42], (const float*)d_in[43],
                                        (const float*)d_in[44], (const float*)d_in[45],
                                        (const float*)d_in[46], b1c);
  concat5_kernel<<<3, 256, 0, stream>>>((const float*)d_in[47], (const float*)d_in[48],
                                        (const float*)d_in[49], (const float*)d_in[50],
                                        (const float*)d_in[51], g2);
  concat5_kernel<<<3, 256, 0, stream>>>((const float*)d_in[52], (const float*)d_in[53],
                                        (const float*)d_in[54], (const float*)d_in[55],
                                        (const float*)d_in[56], b2c);

  // ---- layer 1 ----
  for (int i = 0; i < 5; ++i)
    attention_kernel<<<dim3(64, 512), 64, 0, stream>>>(
        rels[i], relC[i], (const float*)d_in[7 + i], adjs,
        Abf + (size_t)i * 512 * 4096, (i == 4) ? 1 : 0);
  for (int i = 0; i < 5; ++i)
    gconv1_kernel<<<512, 256, 0, stream>>>(
        Abf + (size_t)i * 512 * 4096, afmT,
        (const float*)d_in[17 + i], (const float*)d_in[22 + i], H, i * 128);
  bn_stats_kernel<<<NGC, 256, 0, stream>>>(H, 32768, NGC, mean, rstd);
  bn_apply_mask_kernel<<<(32768 * NGC) / 256, 256, 0, stream>>>(
      H, mean, rstd, g1, b1c, mbuf, x1t, NGC);

  // ---- layer 2 ----
  for (int i = 0; i < 5; ++i)
    attention_kernel<<<dim3(64, 512), 64, 0, stream>>>(
        rels[i], relC[i], (const float*)d_in[12 + i], adjs,
        Abf + (size_t)i * 512 * 4096, 0);
  for (int i = 0; i < 5; ++i)
    gconv2_kernel<<<512, 256, 0, stream>>>(
        Abf + (size_t)i * 512 * 4096, x1t,
        (const float*)d_in[27 + i], (const float*)d_in[32 + i], H, i * 128);
  bn_stats_kernel<<<NGC, 256, 0, stream>>>(H, 32768, NGC, mean, rstd);
  bn_apply_sum_kernel<<<(512 * NGC) / 256, 256, 0, stream>>>(
      H, mean, rstd, g2, b2c, mbuf, molfp, NGC);

  // ---- dense head ----
  bn1d_kernel<<<NGC, 256, 0, stream>>>(molfp, (const float*)d_in[57],
                                       (const float*)d_in[58], xb0, 512, NGC, 0);
  gemm_kernel<<<dim3(4, 8), 256, 0, stream>>>(xb0, (const float*)d_in[59],
                                              (const float*)d_in[60], y1, 512, NGC);
  bn1d_kernel<<<512, 256, 0, stream>>>(y1, (const float*)d_in[61],
                                       (const float*)d_in[62], xb1, 512, 512, 1);
  gemm_kernel<<<dim3(2, 8), 256, 0, stream>>>(xb1, (const float*)d_in[63],
                                              (const float*)d_in[64], y2, 256, 512);
  bn1d_kernel<<<256, 256, 0, stream>>>(y2, (const float*)d_in[65],
                                       (const float*)d_in[66], xb2, 512, 256, 1);
  dense3_kernel<<<512, 256, 0, stream>>>(xb2, (const float*)d_in[67],
                                         (const float*)d_in[68], (float*)d_out);
}
